// MultiHeadAttention_29489245455138
// MI455X (gfx1250) — matrix operand order repaired and checked by reading, not compiled, not run on hardware
//
#include <hip/hip_runtime.h>

// ---------------------------------------------------------------------------
// MI455X (gfx1250) multi-head attention: LN -> QKV proj -> RoPE -> flash attn
// -> out proj.  Matmuls: v_wmma_f32_16x16x32_f16.  Staging: async global->LDS
// (global_load_async_to_lds_b128, ASYNCcnt) with double buffering.  All WMMA
// fragments are loaded with contiguous 16B LDS reads (B operands kept
// transposed end-to-end: Wt in global, Vt in global).
// ---------------------------------------------------------------------------

typedef __attribute__((ext_vector_type(16))) _Float16 v16h;
typedef __attribute__((ext_vector_type(8)))  float    v8f;

static constexpr int Bn   = 4;
static constexpr int Sn   = 2048;
static constexpr int Hn   = 16;
static constexpr int Dh   = 64;
static constexpr int DM   = 1024;
static constexpr int ROWS = Bn * Sn;   // 8192

// ---------------------------------------------------------------------------
// Async global->LDS copy (16 bytes per lane), guarded.
// ---------------------------------------------------------------------------
#if defined(__HIP_DEVICE_COMPILE__) && __has_builtin(__builtin_amdgcn_global_load_async_to_lds_b128)
#define ASYNC_LDS 1
typedef int v4i_t __attribute__((vector_size(16)));
typedef __attribute__((address_space(1))) v4i_t* gp128_t;  // global v4i*
typedef __attribute__((address_space(3))) v4i_t* lp128_t;  // LDS v4i*
__device__ inline void cp16_async(void* lds, const void* g) {
  __builtin_amdgcn_global_load_async_to_lds_b128((gp128_t)g, (lp128_t)lds, 0, 0);
}
template <int N>
__device__ inline void wait_async() {
  __builtin_amdgcn_s_wait_asynccnt(N);   // immediate required
}
#else
#define ASYNC_LDS 0
__device__ inline void cp16_async(void* lds, const void* g) {
  *(uint4*)lds = *(const uint4*)g;
}
template <int N>
__device__ inline void wait_async() {}
#endif

// ---------------------------------------------------------------------------
// WMMA helpers (CDNA5 wave32 layouts, ISA 7.12.2)
// ---------------------------------------------------------------------------
__device__ inline v8f wmma_f16(v16h a, v16h b, v8f c) {
  return __builtin_amdgcn_wmma_f32_16x16x32_f16(false, a, false, b, (short)0, c,
                                                false, false);
}

union frag_u {
  v16h v;
  uint4 q[2];
};

// A fragment: 16x32 f16 from row-major LDS (stride in f16, 16B-aligned rows).
// lane: m=lane&15, half=lane>>4.  VGPR group g holds halves
// rp[g*16 + half*8 .. +7]  ->  two contiguous 16B LDS loads.
__device__ inline v16h load_a_frag(const _Float16* smem, int stride,
                                   int row_base, int k_base, int lane) {
  int m = lane & 15, half = lane >> 4;
  const _Float16* rp =
      smem + (unsigned)(row_base + m) * stride + k_base + half * 8;
  frag_u u;
  u.q[0] = *(const uint4*)(rp);
  u.q[1] = *(const uint4*)(rp + 16);
  return u.v;
}

// B fragment from LDS holding B transposed: Bt[n][k] (stride in f16).
// lane: n=lane&15, half=lane>>4.  VGPR group g holds halves
// rp[g*16 + half*8 .. +7]  ->  two contiguous 16B LDS loads.
__device__ inline v16h load_b_frag_t(const _Float16* smem, int stride,
                                     int n_base, int k_base, int lane) {
  int n = lane & 15, half = lane >> 4;
  const _Float16* rp =
      smem + (unsigned)(n_base + n) * stride + k_base + half * 8;
  frag_u u;
  u.q[0] = *(const uint4*)(rp);
  u.q[1] = *(const uint4*)(rp + 16);
  return u.v;
}

// ---------------------------------------------------------------------------
// Weight convert + transpose: W[k][n] f32 -> Wt[n][k] f16 (LDS-tiled).
// Grid (DM/32, DM/32), block 256 = 32x8.
// ---------------------------------------------------------------------------
__global__ __launch_bounds__(256) void cvt_wt_f16(const float* __restrict__ src,
                                                  _Float16* __restrict__ dst) {
  __shared__ float tile[32][33];
  int tx = threadIdx.x & 31, ty = threadIdx.x >> 5;
  int k0 = blockIdx.x * 32;
  int n0 = blockIdx.y * 32;
#pragma unroll
  for (int i = 0; i < 4; ++i)
    tile[ty + 8 * i][tx] = src[(unsigned)(k0 + ty + 8 * i) * DM + n0 + tx];
  __syncthreads();
#pragma unroll
  for (int i = 0; i < 4; ++i)
    dst[(unsigned)(n0 + ty + 8 * i) * DM + k0 + tx] =
        (_Float16)tile[tx][ty + 8 * i];
}

// ---------------------------------------------------------------------------
// LayerNorm over 1024 cols, one block per row, f16 output
// ---------------------------------------------------------------------------
__global__ __launch_bounds__(256) void ln_f16(const float* __restrict__ x,
                                              const float* __restrict__ g,
                                              const float* __restrict__ bb,
                                              _Float16* __restrict__ y) {
  int row = blockIdx.x;
  const float* xr = x + (unsigned)row * DM;
  __shared__ float red[256];
  int t = threadIdx.x;

  float s = 0.f;
  for (int c = t; c < DM; c += 256) s += xr[c];
  red[t] = s;
  __syncthreads();
  for (int off = 128; off > 0; off >>= 1) {
    if (t < off) red[t] += red[t + off];
    __syncthreads();
  }
  float mu = red[0] * (1.0f / DM);
  __syncthreads();

  float vs = 0.f;
  for (int c = t; c < DM; c += 256) {
    float d = xr[c] - mu;
    vs += d * d;
  }
  red[t] = vs;
  __syncthreads();
  for (int off = 128; off > 0; off >>= 1) {
    if (t < off) red[t] += red[t + off];
    __syncthreads();
  }
  float inv = rsqrtf(red[0] * (1.0f / DM) + 1e-5f);

  _Float16* yr = y + (unsigned)row * DM;
  for (int c = t; c < DM; c += 256)
    yr[c] = (_Float16)((xr[c] - mu) * inv * g[c] + bb[c]);
}

// ---------------------------------------------------------------------------
// GEMM: C = A[M,K] * Wt[N,K]^T + bias.  f16 in, f32 accum, OutT out.
// 256 threads (8 waves), tile 128x64, K-step 32, double-buffered async LDS.
// TOUT=false: C[row][col] (pitch N).
// TOUT=true : C[(b*1024 + col)][s] (pitch Sn), row = b*Sn + s  (V transposed).
// ---------------------------------------------------------------------------
template <typename OutT, bool TOUT>
__global__ __launch_bounds__(256) void gemm_wmma(const _Float16* __restrict__ A,
                                                 const _Float16* __restrict__ Wt,
                                                 const float* __restrict__ bias,
                                                 OutT* __restrict__ C,
                                                 int M, int N, int K) {
  __shared__ _Float16 As[2][128][40];   // 128 rows x 32 k + pad
  __shared__ _Float16 Bts[2][64][40];   // 64 n-rows x 32 k + pad

  int tid  = threadIdx.x;
  int lane = tid & 31;
  int wave = tid >> 5;
  int wm   = wave >> 1;
  int wn   = wave & 1;
  int rowBase = blockIdx.y * 128;
  int colBase = blockIdx.x * 64;

  // staging coords: A 128x32 (2 x b128/thread), Bt 64x32 (1 x b128/thread)
  int ar = tid >> 1, ac = (tid & 1) * 16;
  int bn = tid >> 2, bkc = (tid & 3) * 8;
  const _Float16* agp = A + (unsigned)(rowBase + ar) * K + ac;
  const _Float16* bgp = Wt + (unsigned)(colBase + bn) * K + bkc;

  v8f acc[2][2] = {};

  auto stage = [&](int buf, int ks) {
    const _Float16* ga = agp + ks * 32;
    cp16_async(&As[buf][ar][ac], ga);
    cp16_async(&As[buf][ar][ac + 8], ga + 8);
    cp16_async(&Bts[buf][bn][bkc], bgp + ks * 32);
  };

  int nk = K >> 5;
  stage(0, 0);
  for (int ks = 0; ks < nk; ++ks) {
    int cur = ks & 1;
    if (ks + 1 < nk) {
      stage(cur ^ 1, ks + 1);
      wait_async<3>();   // stage ks landed; ks+1 still in flight
    } else {
      wait_async<0>();
    }
    __syncthreads();

    v16h a0 = load_a_frag(&As[cur][0][0], 40, wm * 32, 0, lane);
    v16h a1 = load_a_frag(&As[cur][0][0], 40, wm * 32 + 16, 0, lane);
    v16h b0 = load_b_frag_t(&Bts[cur][0][0], 40, wn * 32, 0, lane);
    v16h b1 = load_b_frag_t(&Bts[cur][0][0], 40, wn * 32 + 16, 0, lane);
    acc[0][0] = wmma_f16(a0, b0, acc[0][0]);
    acc[0][1] = wmma_f16(a0, b1, acc[0][1]);
    acc[1][0] = wmma_f16(a1, b0, acc[1][0]);
    acc[1][1] = wmma_f16(a1, b1, acc[1][1]);
    __syncthreads();
  }

  // Epilogue: lane n = lane&15, row m = v + 8*(lane>>4)
  int half = lane >> 4;
#pragma unroll
  for (int mt = 0; mt < 2; ++mt) {
#pragma unroll
    for (int nt = 0; nt < 2; ++nt) {
      unsigned col  = colBase + wn * 32 + nt * 16 + (lane & 15);
      unsigned row0 = rowBase + wm * 32 + mt * 16 + 8 * half;
      float bv = bias[col];
      if constexpr (TOUT) {
        // V transposed: out[(b*1024 + col)][s]; tile never straddles a batch
        unsigned bb = row0 >> 11, ss = row0 & (Sn - 1);
        OutT* p = C + (bb * (unsigned)DM + col) * (unsigned)Sn + ss;
#pragma unroll
        for (int v = 0; v < 8; ++v) p[v] = (OutT)(acc[mt][nt][v] + bv);
      } else {
        OutT* p = C + row0 * (unsigned)N + col;
#pragma unroll
        for (int v = 0; v < 8; ++v)
          p[(unsigned)v * (unsigned)N] = (OutT)(acc[mt][nt][v] + bv);
      }
    }
  }
}

// ---------------------------------------------------------------------------
// RoPE (interleaved pairs), in-place on f16 [ROWS][1024], head dim 64.
// ---------------------------------------------------------------------------
__global__ __launch_bounds__(256) void rope_f16(_Float16* __restrict__ buf,
                                                int n_pairs) {
  int i = blockIdx.x * 256 + threadIdx.x;
  if (i >= n_pairs) return;
  unsigned row = (unsigned)i >> 9;
  unsigned col = ((unsigned)i & 511u) << 1;
  int d  = col & (Dh - 1);
  int fi = d >> 1;
  int s  = row & (Sn - 1);
  float freq = __expf(-(float)(2 * fi) * (1.0f / 64.0f) * 9.210340372f);
  float ang  = (float)s * freq;
  float sn, cs;
  __sincosf(ang, &sn, &cs);
  unsigned base = row * DM + col;
  float x1 = (float)buf[base];
  float x2 = (float)buf[base + 1];
  buf[base]     = (_Float16)(x1 * cs - x2 * sn);
  buf[base + 1] = (_Float16)(x2 * cs + x1 * sn);
}

// ---------------------------------------------------------------------------
// Flash attention: block = (b, h, 64 q-rows), 128 threads = 4 waves.
// Double-buffered async K/Vt staging; online softmax; WMMA for QK^T and PV.
// V comes in transposed: Vt[(b*1024 + h*64 + d)][s].
// ---------------------------------------------------------------------------
__global__ __launch_bounds__(128) void attn_wmma(const _Float16* __restrict__ Q,
                                                 const _Float16* __restrict__ K,
                                                 const _Float16* __restrict__ Vt,
                                                 const int* __restrict__ lens,
                                                 _Float16* __restrict__ O) {
  __shared__ _Float16 Qs[64][64];
  __shared__ _Float16 Ks[2][64][64];      // [key][d]
  __shared__ _Float16 Vts[2][64][72];     // [d][key] (transposed, padded)
  __shared__ _Float16 Ps[64][64];
  __shared__ float    Ss[64][68];
  __shared__ float    mrun[64], lrun[64], corr[64];

  int qb = blockIdx.x, h = blockIdx.y, b = blockIdx.z;
  int tid = threadIdx.x, lane = tid & 31, wave = tid >> 5;
  int len = lens[b];
  int qRow0 = qb * 64;

  int r = tid >> 1, c = (tid & 1) * 32;  // staging coords: 32 f16 per thread
  unsigned gq = (unsigned)(b * Sn + qRow0 + r) * DM + h * Dh + c;

  // Q staging (4 x b128 per thread)
#pragma unroll
  for (int i = 0; i < 4; ++i) cp16_async(&Qs[r][c + 8 * i], Q + gq + 8 * i);

  // K/V stage for key block kb (4 + 4 b128 per thread)
  auto stageKV = [&](int buf, int kb) {
    unsigned gk = (unsigned)(b * Sn + kb * 64 + r) * DM + h * Dh + c;
#pragma unroll
    for (int i = 0; i < 4; ++i)
      cp16_async(&Ks[buf][r][c + 8 * i], K + gk + 8 * i);
    // Vt rows are head-dims: row d = r, cols are keys
    unsigned gv = (unsigned)(b * DM + h * Dh + r) * Sn + kb * 64 + c;
#pragma unroll
    for (int i = 0; i < 4; ++i)
      cp16_async(&Vts[buf][r][c + 8 * i], Vt + gv + 8 * i);
  };

  int nkb = (len + 63) >> 6;
  stageKV(0, 0);

  if (tid < 64) {
    mrun[tid] = -3.4e38f;
    lrun[tid] = 0.f;
  }

  v8f oacc[4] = {};
  int half = lane >> 4;

  for (int kb = 0; kb < nkb; ++kb) {
    int cur = kb & 1;
    if (kb + 1 < nkb) {
      stageKV(cur ^ 1, kb + 1);
      wait_async<8>();   // in-order: Q + stage kb landed, kb+1 in flight
    } else {
      wait_async<0>();
    }
    __syncthreads();

    v16h qa0 = load_a_frag(&Qs[0][0], 64, wave * 16, 0, lane);
    v16h qa1 = load_a_frag(&Qs[0][0], 64, wave * 16, 32, lane);

    // scores: S = Q*K^T * scale   (B[k][n] = Ks[n][k] -> transposed frag)
#pragma unroll
    for (int nt = 0; nt < 4; ++nt) {
      v16h kb0 = load_b_frag_t(&Ks[cur][0][0], 64, nt * 16, 0, lane);
      v16h kb1 = load_b_frag_t(&Ks[cur][0][0], 64, nt * 16, 32, lane);
      v8f sacc = {};
      sacc = wmma_f16(qa0, kb0, sacc);
      sacc = wmma_f16(qa1, kb1, sacc);
#pragma unroll
      for (int v = 0; v < 8; ++v)
        Ss[wave * 16 + v + 8 * half][nt * 16 + (lane & 15)] = sacc[v] * 0.125f;
    }
    __syncthreads();

    // online softmax over this 64-wide key block (threads 0..63: one row each)
    if (tid < 64) {
      int rr = tid;
      int vlim = len - kb * 64;
      if (vlim > 64) vlim = 64;
      float bm = -3.4e38f;
      for (int j = 0; j < vlim; ++j) bm = fmaxf(bm, Ss[rr][j]);
      float nm = fmaxf(mrun[rr], bm);
      float cf = __expf(mrun[rr] - nm);
      float bsum = 0.f;
      for (int j = 0; j < 64; ++j) {
        float p = (j < vlim) ? __expf(Ss[rr][j] - nm) : 0.f;
        Ps[rr][j] = (_Float16)p;
        bsum += p;
      }
      lrun[rr] = lrun[rr] * cf + bsum;
      mrun[rr] = nm;
      corr[rr] = cf;
    }
    __syncthreads();

    // rescale running O, then O += P * V
    float cf[8];
#pragma unroll
    for (int v = 0; v < 8; ++v) cf[v] = corr[wave * 16 + v + 8 * half];
#pragma unroll
    for (int nt = 0; nt < 4; ++nt)
#pragma unroll
      for (int v = 0; v < 8; ++v) oacc[nt][v] *= cf[v];

    v16h pa0 = load_a_frag(&Ps[0][0], 64, wave * 16, 0, lane);
    v16h pa1 = load_a_frag(&Ps[0][0], 64, wave * 16, 32, lane);
    // PV: B[k=key][n=d] = Vts[d][key] -> transposed frag, contiguous pairs
#pragma unroll
    for (int nt = 0; nt < 4; ++nt) {
      v16h vb0 = load_b_frag_t(&Vts[cur][0][0], 72, nt * 16, 0, lane);
      v16h vb1 = load_b_frag_t(&Vts[cur][0][0], 72, nt * 16, 32, lane);
      oacc[nt] = wmma_f16(pa0, vb0, oacc[nt]);
      oacc[nt] = wmma_f16(pa1, vb1, oacc[nt]);
    }
    __syncthreads();  // all reads of buf `cur` done before it is re-staged
  }

  // finalize: divide by row sum, write f16 [b*S+q][h*64+d]
  float denom[8];
#pragma unroll
  for (int v = 0; v < 8; ++v) denom[v] = lrun[wave * 16 + v + 8 * half];
#pragma unroll
  for (int nt = 0; nt < 4; ++nt) {
    unsigned col = h * Dh + nt * 16 + (lane & 15);
    unsigned row0 = (unsigned)(b * Sn + qRow0 + wave * 16 + 8 * half);
    _Float16* p = O + row0 * DM + col;
#pragma unroll
    for (int v = 0; v < 8; ++v)
      p[(unsigned)v * DM] = (_Float16)(oacc[nt][v] / denom[v]);
  }
}

// ---------------------------------------------------------------------------
// Host launcher
// ---------------------------------------------------------------------------
extern "C" void kernel_launch(void* const* d_in, const int* in_sizes, int n_in,
                              void* d_out, int out_size, void* d_ws,
                              size_t ws_size, hipStream_t stream) {
  (void)in_sizes; (void)n_in; (void)out_size; (void)ws_size;

  const float* xq   = (const float*)d_in[0];
  const float* xk   = (const float*)d_in[1];
  const float* xv   = (const float*)d_in[2];
  const int*   lens = (const int*)d_in[3];
  const float* lng  = (const float*)d_in[4];
  const float* lnb  = (const float*)d_in[5];
  const float* Wq   = (const float*)d_in[6];
  const float* bq   = (const float*)d_in[7];
  const float* Wk   = (const float*)d_in[8];
  const float* bk   = (const float*)d_in[9];
  const float* Wv   = (const float*)d_in[10];
  const float* bv   = (const float*)d_in[11];
  const float* Wo   = (const float*)d_in[12];
  const float* bo   = (const float*)d_in[13];
  float* out = (float*)d_out;

  char* w = (char*)d_ws;
  const size_t act = (size_t)ROWS * DM * sizeof(_Float16);  // 16 MB
  const size_t wsz = (size_t)DM * DM * sizeof(_Float16);    // 2 MB
  _Float16* XLNq = (_Float16*)(w + 0 * act);
  _Float16* XLNk = (_Float16*)(w + 1 * act);
  _Float16* XLNv = (_Float16*)(w + 2 * act);
  _Float16* Qf   = (_Float16*)(w + 3 * act);
  _Float16* Kf   = (_Float16*)(w + 4 * act);
  _Float16* Vt   = (_Float16*)(w + 5 * act);   // transposed V
  _Float16* AO   = (_Float16*)(w + 6 * act);
  _Float16* Wq16 = (_Float16*)(w + 7 * act + 0 * wsz);   // all transposed
  _Float16* Wk16 = (_Float16*)(w + 7 * act + 1 * wsz);
  _Float16* Wv16 = (_Float16*)(w + 7 * act + 2 * wsz);
  _Float16* Wo16 = (_Float16*)(w + 7 * act + 3 * wsz);

  dim3 t256(256);
  dim3 cvtG(DM / 32, DM / 32);
  cvt_wt_f16<<<cvtG, t256, 0, stream>>>(Wq, Wq16);
  cvt_wt_f16<<<cvtG, t256, 0, stream>>>(Wk, Wk16);
  cvt_wt_f16<<<cvtG, t256, 0, stream>>>(Wv, Wv16);
  cvt_wt_f16<<<cvtG, t256, 0, stream>>>(Wo, Wo16);

  ln_f16<<<dim3(ROWS), t256, 0, stream>>>(xq, lng, lnb, XLNq);
  ln_f16<<<dim3(ROWS), t256, 0, stream>>>(xk, lng, lnb, XLNk);
  ln_f16<<<dim3(ROWS), t256, 0, stream>>>(xv, lng, lnb, XLNv);

  dim3 gemmG(DM / 64, ROWS / 128);
  gemm_wmma<_Float16, false><<<gemmG, t256, 0, stream>>>(XLNq, Wq16, bq, Qf,
                                                         ROWS, DM, DM);
  gemm_wmma<_Float16, false><<<gemmG, t256, 0, stream>>>(XLNk, Wk16, bk, Kf,
                                                         ROWS, DM, DM);
  gemm_wmma<_Float16, true><<<gemmG, t256, 0, stream>>>(XLNv, Wv16, bv, Vt,
                                                        ROWS, DM, DM);

  const int npairs = ROWS * (DM / 2);
  dim3 ropeG((npairs + 255) / 256);
  rope_f16<<<ropeG, t256, 0, stream>>>(Qf, npairs);
  rope_f16<<<ropeG, t256, 0, stream>>>(Kf, npairs);

  attn_wmma<<<dim3(Sn / 64, Hn, Bn), dim3(128), 0, stream>>>(Qf, Kf, Vt, lens, AO);

  gemm_wmma<float, false><<<gemmG, t256, 0, stream>>>(AO, Wo16, bo, out,
                                                      ROWS, DM, DM);
}
